// MeshDeformationBlock_88021059764779
// MI455X (gfx1250) — matrix operand order repaired and checked by reading, not compiled, not run on hardware
//
#include <hip/hip_runtime.h>
#include <hip/hip_bf16.h>
#include <stdint.h>

#define B_ 16
#define V_ 40000
#define C_ 128
#define H_ 56
#define W_ 56
#define E_ 120000
#define HW_ (H_ * W_)

typedef __attribute__((ext_vector_type(16))) __bf16 v16bf;
typedef __attribute__((ext_vector_type(8)))  float  v8f;

// ---------------------------------------------------------------------------
// Weight swizzle: f32 [128x128] row-major -> bf16 fragments in WMMA B layout.
// Fragment order: [ntile(8)][kblock(4)][lane(32)][elem(16)]
//   element(lane,i) = W[k][n], k = kblock*32 + (i>>3)*16 + (lane>>4)*8 + (i&7),
//                              n = ntile*16 + (lane&15)
// ---------------------------------------------------------------------------
__global__ void swizzle_weights_kernel(const float* __restrict__ w,
                                       __bf16* __restrict__ dst) {
  int p    = blockIdx.x * blockDim.x + threadIdx.x;  // 0..16383
  int i    = p & 15;
  int lane = (p >> 4) & 31;
  int kb   = (p >> 9) & 3;
  int nt   = p >> 11;
  int k = kb * 32 + ((i >> 3) << 4) + ((lane >> 4) << 3) + (i & 7);
  int n = nt * 16 + (lane & 15);
  dst[p] = (__bf16)w[k * C_ + n];
}

// ---------------------------------------------------------------------------
// Transpose img_features [B,C,H*W] -> [B,H*W,C] so vert_align taps are
// contiguous 128-float rows. 32x32 LDS tile (pad 33 to dodge bank conflicts).
// ---------------------------------------------------------------------------
__global__ void transpose_feats_kernel(const float* __restrict__ in,
                                       float* __restrict__ out) {
  __shared__ float tile[32][33];
  int b  = blockIdx.z;
  int c0 = blockIdx.y * 32;        // channel tile base
  int p0 = blockIdx.x * 32;        // hw tile base (3136 = 98*32)
  int tx = threadIdx.x;            // 0..31
  int ty = threadIdx.y;            // 0..7
  const float* inb  = in  + (size_t)b * C_ * HW_;
  float*       outb = out + (size_t)b * HW_ * C_;
#pragma unroll
  for (int r = 0; r < 32; r += 8)
    tile[ty + r][tx] = inb[(size_t)(c0 + ty + r) * HW_ + p0 + tx];
  __syncthreads();
#pragma unroll
  for (int r = 0; r < 32; r += 8)
    outb[(size_t)(p0 + ty + r) * C_ + c0 + tx] = tile[tx][ty + r];
}

// ---------------------------------------------------------------------------
// vert_align (bilinear, align_corners, zero OOB) + vertex_padded.
// feats [B,H,W,C]: each tap is a coalesced 128-float row. 2 vertices/block.
// ---------------------------------------------------------------------------
__global__ void vert_align_kernel(const float* __restrict__ featsT,
                                  const float* __restrict__ verts,
                                  const float* __restrict__ vpad,
                                  float* __restrict__ g) {
  int bv = blockIdx.x * 2 + (threadIdx.x >> 7);   // b*V + v
  int b  = bv / V_;
  int c  = threadIdx.x & 127;
  const float* vp = verts + (size_t)bv * 3;
  float x = (vp[0] + 1.f) * 0.5f * (float)(W_ - 1);
  float y = (vp[1] + 1.f) * 0.5f * (float)(H_ - 1);
  float x0f = floorf(x), y0f = floorf(y);
  float wx1 = x - x0f, wx0 = 1.f - wx1;
  float wy1 = y - y0f, wy0 = 1.f - wy1;
  int x0 = (int)x0f, y0 = (int)y0f;
  const float* fb = featsT + (size_t)b * HW_ * C_;
  float acc = 0.f;
#pragma unroll
  for (int ty = 0; ty < 2; ++ty) {
    int   yi = y0 + ty;
    float wy = ty ? wy1 : wy0;
#pragma unroll
    for (int tx = 0; tx < 2; ++tx) {
      int   xi = x0 + tx;
      float wx = tx ? wx1 : wx0;
      bool inb = (xi >= 0) & (xi < W_) & (yi >= 0) & (yi < H_);
      int  xc = xi < 0 ? 0 : (xi > W_ - 1 ? W_ - 1 : xi);
      int  yc = yi < 0 ? 0 : (yi > H_ - 1 ? H_ - 1 : yi);
      acc += fb[(size_t)(yc * W_ + xc) * C_ + c] * (inb ? wx * wy : 0.f);
    }
  }
  g[(size_t)bv * C_ + c] = acc + vpad[(size_t)bv * C_ + c];
}

// ---------------------------------------------------------------------------
// Fused dual-GEMM, v4: block = 8 waves; each wave owns TWO 16-row A tiles
// (32 rows), so every LDS B fragment feeds two independent WMMA chains
// (acc0/acc1) — halves ds traffic per FLOP and fills WMMA RAW-hazard slots.
// Block covers 256 rows; weights staged once in 64 KB LDS; bias in registers.
// 128 x v_wmma_f32_16x16x32_bf16 per wave.
// ---------------------------------------------------------------------------
__global__ void __launch_bounds__(256)
graph_gemm_kernel(const float* __restrict__ X,
                  const __bf16* __restrict__ w0swz,
                  const __bf16* __restrict__ w1swz,
                  const float* __restrict__ bias,
                  float* __restrict__ outBase,
                  float* __restrict__ outY1) {
  __shared__ __bf16 ldsw[2 * 16384];  // [mat][frag-order], 64 KB

  const int lane = threadIdx.x & 31;
  const int wave = threadIdx.x >> 5;
  const int m0   = (blockIdx.x * 8 + wave) * 32;
  const int half = lane >> 4;

  // Issue both A-tile loads (16 x b128 per lane) + bias loads up front.
  float4 fa[2][4][4];
#pragma unroll
  for (int t = 0; t < 2; ++t) {
    const float* xrow = X + (size_t)(m0 + t * 16 + (lane & 15)) * C_;
#pragma unroll
    for (int kb = 0; kb < 4; ++kb) {
      const float4* p0 = (const float4*)(xrow + kb * 32 + half * 8);
      const float4* p1 = (const float4*)(xrow + kb * 32 + 16 + half * 8);
      fa[t][kb][0] = p0[0]; fa[t][kb][1] = p0[1];
      fa[t][kb][2] = p1[0]; fa[t][kb][3] = p1[1];
    }
  }
  float biasv[8];
#pragma unroll
  for (int nt = 0; nt < 8; ++nt) biasv[nt] = bias[nt * 16 + (lane & 15)];

  // Cooperative 64 KB weight staging (overlaps the loads above).
  {
    const uint4* s0 = (const uint4*)w0swz;
    const uint4* s1 = (const uint4*)w1swz;
    uint4* dst = (uint4*)ldsw;
    for (int t = threadIdx.x; t < 2048; t += 256) {
      dst[t]        = s0[t];
      dst[t + 2048] = s1[t];
    }
  }

  // f32 -> bf16 A fragments (16-bit A 16x32 layout, ISA 7.12.2).
  v16bf a[2][4];
#pragma unroll
  for (int t = 0; t < 2; ++t)
#pragma unroll
    for (int kb = 0; kb < 4; ++kb) {
      v16bf av;
#pragma unroll
      for (int q = 0; q < 4; ++q) {
        av[q * 4 + 0] = (__bf16)fa[t][kb][q].x;
        av[q * 4 + 1] = (__bf16)fa[t][kb][q].y;
        av[q * 4 + 2] = (__bf16)fa[t][kb][q].z;
        av[q * 4 + 3] = (__bf16)fa[t][kb][q].w;
      }
      a[t][kb] = av;
    }
  __syncthreads();

#pragma unroll
  for (int mat = 0; mat < 2; ++mat) {
    const __bf16* wl   = ldsw + mat * 16384;
    float*        outp = mat ? outY1 : outBase;

    // Double-buffered B fragments.
    v16bf bf[2][4];
#pragma unroll
    for (int kb = 0; kb < 4; ++kb)
      bf[0][kb] = *(const v16bf*)(wl + ((kb * 32 + lane) << 4));

#pragma unroll
    for (int nt = 0; nt < 8; ++nt) {
      const int cur = nt & 1;
      const int nxt = cur ^ 1;
      if (nt < 7) {
#pragma unroll
        for (int kb = 0; kb < 4; ++kb)
          bf[nxt][kb] =
              *(const v16bf*)(wl + ((((nt + 1) * 4 + kb) * 32 + lane) << 4));
      }
      v8f acc0 = {0.f, 0.f, 0.f, 0.f, 0.f, 0.f, 0.f, 0.f};
      v8f acc1 = {0.f, 0.f, 0.f, 0.f, 0.f, 0.f, 0.f, 0.f};
#pragma unroll
      for (int kb = 0; kb < 4; ++kb) {
        acc0 = __builtin_amdgcn_wmma_f32_16x16x32_bf16(
            false, a[0][kb], false, bf[cur][kb], (short)0, acc0, false, false);
        acc1 = __builtin_amdgcn_wmma_f32_16x16x32_bf16(
            false, a[1][kb], false, bf[cur][kb], (short)0, acc1, false, false);
      }
      int   col = nt * 16 + (lane & 15);
      float bv  = mat ? 0.f : biasv[nt];
#pragma unroll
      for (int r = 0; r < 8; ++r) {
        outp[(size_t)(m0 + half * 8 + r) * C_ + col]      = acc0[r] + bv;
        outp[(size_t)(m0 + 16 + half * 8 + r) * C_ + col] = acc1[r] + bv;
      }
    }
  }
}

// ---------------------------------------------------------------------------
// Undirected-edge scatter-add via hardware global_atomic_add_f32.
// 256-thread blocks: threads 0-127 -> batch 2*bp, 128-255 -> batch 2*bp+1.
// ---------------------------------------------------------------------------
__global__ void scatter_add_kernel(const int* __restrict__ edges,
                                   const float* __restrict__ y1,
                                   float* __restrict__ out) {
  int idx = blockIdx.x;                 // e + E_*bpair
  int e   = idx % E_;
  int bp  = idx / E_;
  int b   = bp * 2 + (threadIdx.x >> 7);
  int c   = threadIdx.x & 127;
  int i   = edges[2 * e];
  int j   = edges[2 * e + 1];
  size_t bi = ((size_t)b * V_ + i) * C_ + c;
  size_t bj = ((size_t)b * V_ + j) * C_ + c;
  float yj = y1[bj];
  float yi = y1[bi];
  __hip_atomic_fetch_add(&out[bi], yj, __ATOMIC_RELAXED, __HIP_MEMORY_SCOPE_AGENT);
  __hip_atomic_fetch_add(&out[bj], yi, __ATOMIC_RELAXED, __HIP_MEMORY_SCOPE_AGENT);
}

// out += a   (out already holds d, a holds d1)
__global__ void final_add_kernel(const float* __restrict__ a,
                                 float* __restrict__ out) {
  size_t t = (size_t)blockIdx.x * blockDim.x + threadIdx.x;
  float4 va = ((const float4*)a)[t];
  float4 vo = ((float4*)out)[t];
  vo.x += va.x; vo.y += va.y; vo.z += va.z; vo.w += va.w;
  ((float4*)out)[t] = vo;
}

extern "C" void kernel_launch(void* const* d_in, const int* in_sizes, int n_in,
                              void* d_out, int out_size, void* d_ws, size_t ws_size,
                              hipStream_t stream) {
  (void)in_sizes; (void)n_in; (void)out_size; (void)ws_size;
  const float* feats = (const float*)d_in[0];
  const float* verts = (const float*)d_in[1];
  const float* vpad  = (const float*)d_in[2];
  const int*   edges = (const int*)d_in[3];
  const float* w0[3] = {(const float*)d_in[4], (const float*)d_in[7],  (const float*)d_in[10]};
  const float* b0[3] = {(const float*)d_in[5], (const float*)d_in[8],  (const float*)d_in[11]};
  const float* w1[3] = {(const float*)d_in[6], (const float*)d_in[9],  (const float*)d_in[12]};

  // Workspace: [6 x 32KB swz weights | pad to 256KB][buf0][buf2][buf3][featsT]
  // buf0 = g, later d1; buf2 = layer-2 out; buf3 = y1 scratch; d lives in d_out.
  char* ws = (char*)d_ws;
  __bf16* wswz[6];
  for (int m = 0; m < 6; ++m)
    wswz[m] = (__bf16*)(ws + (size_t)m * 16384 * sizeof(__bf16));
  const size_t bufElems = (size_t)B_ * V_ * C_;
  float* buf0   = (float*)(ws + (size_t)(1 << 18));
  float* buf2   = buf0 + bufElems;
  float* buf3   = buf2 + bufElems;
  float* featsT = buf3 + bufElems;          // 25.7 MB
  float* bufD   = (float*)d_out;

  for (int l = 0; l < 3; ++l) {
    swizzle_weights_kernel<<<64, 256, 0, stream>>>(w0[l], wswz[2 * l]);
    swizzle_weights_kernel<<<64, 256, 0, stream>>>(w1[l], wswz[2 * l + 1]);
  }

  transpose_feats_kernel<<<dim3(HW_ / 32, C_ / 32, B_), dim3(32, 8), 0, stream>>>(feats, featsT);
  vert_align_kernel<<<(B_ * V_) / 2, 256, 0, stream>>>(featsT, verts, vpad, buf0);

  const int gemmGrid = (B_ * V_) / 256;     // 2500 blocks, 8 waves x 32 rows
  // layer 1: g -> d (in d_out)
  graph_gemm_kernel<<<gemmGrid, 256, 0, stream>>>(buf0, wswz[0], wswz[1], b0[0], bufD, buf3);
  scatter_add_kernel<<<E_ * (B_ / 2), 256, 0, stream>>>(edges, buf3, bufD);
  // layer 2: d -> buf2
  graph_gemm_kernel<<<gemmGrid, 256, 0, stream>>>(bufD, wswz[2], wswz[3], b0[1], buf2, buf3);
  scatter_add_kernel<<<E_ * (B_ / 2), 256, 0, stream>>>(edges, buf3, buf2);
  // layer 3: buf2 -> d1 (reuse buf0)
  graph_gemm_kernel<<<gemmGrid, 256, 0, stream>>>(buf2, wswz[4], wswz[5], b0[2], buf0, buf3);
  scatter_add_kernel<<<E_ * (B_ / 2), 256, 0, stream>>>(edges, buf3, buf0);
  // out = d + d1
  final_add_kernel<<<(int)(bufElems / 4 / 256), 256, 0, stream>>>(buf0, bufD);
}
